// SSMBlock_76639396429888
// MI455X (gfx1250) — compile-verified, matrix operand order checked by reading, not run on hardware
//
#include <hip/hip_runtime.h>
#include <hip/hip_bf16.h>
#include <math.h>

#define DIMX   1024
#define STATE  16
#define INNER  2048
#define NPROJ  4096
#define KWK    4
#define BB     4
#define LLEN   2048
#define MROWS  (BB * LLEN)   // 8192

typedef __bf16 v16bf __attribute__((ext_vector_type(16)));
typedef float  v8f   __attribute__((ext_vector_type(8)));

struct Frag32 { uint4 lo, hi; };

__device__ __forceinline__ unsigned short f2bf(float f) {
    unsigned u = __builtin_bit_cast(unsigned, f);
    unsigned r = 0x7FFFu + ((u >> 16) & 1u);
    return (unsigned short)((u + r) >> 16);
}

// Lane fragment load: p already points at (row_ptr + half*8).
// ISA 16-bit operand layout: K chunks [hl*8, hl*8+8) and [16+hl*8, +8).
__device__ __forceinline__ v16bf load_frag(const unsigned short* p) {
    Frag32 f;
    f.lo = *(const uint4*)(p);
    f.hi = *(const uint4*)(p + 16);
    return __builtin_bit_cast(v16bf, f);
}

// Async copy of 32 contiguous bytes global -> LDS (2x b128, ASYNCcnt).
// INST_OFFSET applies to both the LDS and global addresses (ISA 10.7).
__device__ __forceinline__ void async_ld32(unsigned lds_off, const void* g) {
    asm volatile(
        "global_load_async_to_lds_b128 %0, %1, off\n\t"
        "global_load_async_to_lds_b128 %0, %1, off offset:16"
        :: "v"(lds_off), "v"(g) : "memory");
}

__device__ __forceinline__ void wait_async0() {
    asm volatile("s_wait_asynccnt 0x0" ::: "memory");
}

// ---------------------------------------------------------------- convert
__global__ __launch_bounds__(256) void cvt_bf16(const float* __restrict__ src,
                                                unsigned short* __restrict__ dst,
                                                int n) {
    int i = blockIdx.x * blockDim.x + threadIdx.x;
    if (i < n) dst[i] = f2bf(src[i]);
}

// ---------------------------------------------------------------- big GEMM
// C[M,N] = A[M,K] * B[N,K]^T ; A,B bf16 row-major (K contiguous), C f32.
// Block tile 128x256, wave tile 64x64 (16 WMMAs / K-step), BK=32,
// double-buffered LDS filled with global_load_async_to_lds_b128.
// mode 0: plain store. mode 1: softplus(v + bias[n]) (dt projection).
#define BM 128
#define BN 256
#define BKSTEP 32
#define LDT    40   // padded LDS row stride (ushorts)

__global__ __launch_bounds__(256)
void gemm_bf16(const unsigned short* __restrict__ A,
               const unsigned short* __restrict__ Bw,
               float* __restrict__ C,
               const float* __restrict__ bias,
               int N, int K, int mode) {
    __shared__ unsigned short sA[2][BM * LDT];
    __shared__ unsigned short sB[2][BN * LDT];

    const int t    = threadIdx.x;
    const int wave = t >> 5;
    const int lane = t & 31;
    const int hl   = lane >> 4;   // lane half (K-chunk select)
    const int r    = lane & 15;   // row-within-fragment
    const int waveM = wave >> 2;  // 0..1 -> 64 rows
    const int waveN = wave & 3;   // 0..3 -> 64 cols
    const int rowBase = blockIdx.y * BM;
    const int colBase = blockIdx.x * BN;

    // staging assignment: A 128 rows x 2 segs = 256 slots (1/thread);
    //                     B 256 rows x 2 segs = 512 slots (2/thread)
    const int ldRow = t >> 1;
    const int seg   = t & 1;
    const unsigned short* gA  = A  + (size_t)(rowBase + ldRow) * K + seg * 16;
    const unsigned short* gB0 = Bw + (size_t)(colBase + ldRow) * K + seg * 16;
    const unsigned short* gB1 = Bw + (size_t)(colBase + 128 + ldRow) * K + seg * 16;

    unsigned aoff[2], boff0[2], boff1[2];
#pragma unroll
    for (int bsel = 0; bsel < 2; ++bsel) {
        aoff[bsel]  = (unsigned)(size_t)&sA[bsel][ldRow * LDT + seg * 16];
        boff0[bsel] = (unsigned)(size_t)&sB[bsel][ldRow * LDT + seg * 16];
        boff1[bsel] = (unsigned)(size_t)&sB[bsel][(128 + ldRow) * LDT + seg * 16];
    }

    v8f acc[4][4] = {};

    // prologue: fill buffer 0
    async_ld32(aoff[0], gA);
    async_ld32(boff0[0], gB0);
    async_ld32(boff1[0], gB1);

    const int nk = K / BKSTEP;
    for (int i = 0; i < nk; ++i) {
        wait_async0();       // my async writes for tile i landed in LDS
        __syncthreads();     // everyone's landed; everyone done with other buf

        if (i + 1 < nk) {    // overlap: fill the other buffer during compute
            gA += BKSTEP; gB0 += BKSTEP; gB1 += BKSTEP;
            const int nb = (i + 1) & 1;
            async_ld32(aoff[nb], gA);
            async_ld32(boff0[nb], gB0);
            async_ld32(boff1[nb], gB1);
        }

        const int cur = i & 1;
        v16bf af[4], bfr[4];
#pragma unroll
        for (int mi = 0; mi < 4; ++mi)
            af[mi] = load_frag(&sA[cur][(waveM * 64 + mi * 16 + r) * LDT + hl * 8]);
#pragma unroll
        for (int ni = 0; ni < 4; ++ni)
            bfr[ni] = load_frag(&sB[cur][(waveN * 64 + ni * 16 + r) * LDT + hl * 8]);
#pragma unroll
        for (int mi = 0; mi < 4; ++mi)
#pragma unroll
            for (int ni = 0; ni < 4; ++ni)
                acc[mi][ni] = __builtin_amdgcn_wmma_f32_16x16x32_bf16(
                    false, af[mi], false, bfr[ni], (short)0, acc[mi][ni],
                    false, false);
    }

#pragma unroll
    for (int mi = 0; mi < 4; ++mi) {
#pragma unroll
        for (int ni = 0; ni < 4; ++ni) {
#pragma unroll
            for (int j = 0; j < 8; ++j) {
                int m = rowBase + waveM * 64 + mi * 16 + hl * 8 + j;
                int n = colBase + waveN * 64 + ni * 16 + r;
                float v = acc[mi][ni][j];
                if (mode == 1) {
                    v += bias[n];
                    v = (v > 20.f) ? v : __logf(1.f + __expf(v));
                }
                C[(size_t)m * N + n] = v;
            }
        }
    }
}

// --------------------------------------------- skinny N=16 GEMM (B/C proj)
// O[M,16] = A[M,K] * W[16,K]^T. blockIdx.y selects (B_w,Bm) vs (C_w,Cm).
__global__ __launch_bounds__(256)
void gemm_bc(const unsigned short* __restrict__ A,
             const unsigned short* __restrict__ W0,
             const unsigned short* __restrict__ W1,
             float* __restrict__ O0, float* __restrict__ O1, int K) {
    const int t = threadIdx.x, wave = t >> 5, lane = t & 31;
    const int hl = lane >> 4, r = lane & 15;
    const unsigned short* W = blockIdx.y ? W1 : W0;
    float* O = blockIdx.y ? O1 : O0;
    const int mBase = blockIdx.x * 128 + wave * 16;

    v8f acc = {};
    const unsigned short* ap = A + (size_t)(mBase + r) * K + hl * 8;
    const unsigned short* bp = W + (size_t)r * K + hl * 8;
    for (int k = 0; k < K; k += 32) {
        v16bf af = load_frag(ap); ap += 32;
        v16bf bf2 = load_frag(bp); bp += 32;
        __builtin_prefetch((const void*)ap, 0, 1);  // global_prefetch_b8
        acc = __builtin_amdgcn_wmma_f32_16x16x32_bf16(
            false, af, false, bf2, (short)0, acc, false, false);
    }
#pragma unroll
    for (int j = 0; j < 8; ++j)
        O[(size_t)(mBase + hl * 8 + j) * STATE + r] = acc[j];
}

// ---------------------------------------------------------- conv + SiLU
__global__ __launch_bounds__(256)
void conv_silu(const float* __restrict__ xr, const float* __restrict__ cw,
               const float* __restrict__ cb, float* __restrict__ xin,
               unsigned short* __restrict__ xin_bf) {
    long long i = (long long)blockIdx.x * blockDim.x + threadIdx.x;
    if (i >= (long long)MROWS * INNER) return;
    int d = (int)(i % INNER);
    long long row = i / INNER;          // b*L + l
    int l = (int)(row % LLEN);
    float acc = cb[d];
#pragma unroll
    for (int k = 0; k < KWK; ++k) {
        int ll = l - 3 + k;
        if (ll >= 0)
            acc += cw[d * KWK + k] * xr[(row + (ll - l)) * (long long)NPROJ + d];
    }
    float s = acc / (1.f + __expf(-acc));
    xin[i] = s;
    xin_bf[i] = f2bf(s);
}

// ---------------------------------------------------------- selective scan
__global__ __launch_bounds__(256)
void scan_kernel(const float* __restrict__ dt, const float* __restrict__ xin,
                 const float* __restrict__ Bm, const float* __restrict__ Cm,
                 const float* __restrict__ Alog, float* __restrict__ y) {
    int d = blockIdx.x * blockDim.x + threadIdx.x;   // 0..INNER-1
    int b = blockIdx.y;

    float a[STATE], st[STATE];
#pragma unroll
    for (int s = 0; s < STATE; ++s) {
        a[s] = -__expf(Alog[d * STATE + s]);
        st[s] = 0.f;
    }
    const float* dtp = dt  + (size_t)b * LLEN * INNER + d;
    const float* xp  = xin + (size_t)b * LLEN * INNER + d;
    const float4* Bp = (const float4*)(Bm + (size_t)b * LLEN * STATE);
    const float4* Cp = (const float4*)(Cm + (size_t)b * LLEN * STATE);
    float* yp = y + (size_t)b * LLEN * INNER + d;

    for (int l = 0; l < LLEN; ++l) {
        float dv = dtp[(size_t)l * INNER];
        float xv = xp[(size_t)l * INNER];
        float dx = dv * xv;
        float yv = 0.f;
#pragma unroll
        for (int q = 0; q < 4; ++q) {
            float4 Bv = Bp[l * 4 + q];
            float4 Cv = Cp[l * 4 + q];
            int s0 = q * 4;
            st[s0 + 0] = __expf(a[s0 + 0] * dv) * st[s0 + 0] + dx * Bv.x;
            st[s0 + 1] = __expf(a[s0 + 1] * dv) * st[s0 + 1] + dx * Bv.y;
            st[s0 + 2] = __expf(a[s0 + 2] * dv) * st[s0 + 2] + dx * Bv.z;
            st[s0 + 3] = __expf(a[s0 + 3] * dv) * st[s0 + 3] + dx * Bv.w;
            yv += st[s0 + 0] * Cv.x + st[s0 + 1] * Cv.y +
                  st[s0 + 2] * Cv.z + st[s0 + 3] * Cv.w;
        }
        yp[(size_t)l * INNER] = yv;
    }
}

// ---------------------------------------------------------- z = y*silu(res)
__global__ __launch_bounds__(256)
void zprep(const float* __restrict__ y, const float* __restrict__ xr,
           unsigned short* __restrict__ z) {
    long long i = (long long)blockIdx.x * blockDim.x + threadIdx.x;
    if (i >= (long long)MROWS * INNER) return;
    long long row = i / INNER;
    int d = (int)(i % INNER);
    float res = xr[row * (long long)NPROJ + INNER + d];
    float v = y[i] * (res / (1.f + __expf(-res)));
    z[i] = f2bf(v);
}

// =======================================================================
extern "C" void kernel_launch(void* const* d_in, const int* in_sizes, int n_in,
                              void* d_out, int out_size, void* d_ws, size_t ws_size,
                              hipStream_t stream) {
    const float* x        = (const float*)d_in[0];
    const float* in_proj  = (const float*)d_in[1];
    const float* conv_w   = (const float*)d_in[2];
    const float* conv_b   = (const float*)d_in[3];
    const float* dt_w     = (const float*)d_in[4];
    const float* dt_b     = (const float*)d_in[5];
    const float* A_log    = (const float*)d_in[6];
    const float* B_w      = (const float*)d_in[7];
    const float* C_w      = (const float*)d_in[8];
    const float* out_w    = (const float*)d_in[9];
    float* out = (float*)d_out;

    char* p = (char*)d_ws;
    auto alloc = [&](size_t bytes) {
        char* r = p;
        p += (bytes + 255) & ~(size_t)255;
        return r;
    };
    unsigned short* x_bf    = (unsigned short*)alloc((size_t)MROWS * DIMX * 2);
    unsigned short* wip_bf  = (unsigned short*)alloc((size_t)NPROJ * DIMX * 2);
    unsigned short* wdt_bf  = (unsigned short*)alloc((size_t)INNER * INNER * 2);
    unsigned short* wout_bf = (unsigned short*)alloc((size_t)DIMX * INNER * 2);
    unsigned short* wB_bf   = (unsigned short*)alloc((size_t)STATE * INNER * 2);
    unsigned short* wC_bf   = (unsigned short*)alloc((size_t)STATE * INNER * 2);
    float*          xr      = (float*)alloc((size_t)MROWS * NPROJ * 4);
    float*          xin     = (float*)alloc((size_t)MROWS * INNER * 4);
    unsigned short* xin_bf  = (unsigned short*)alloc((size_t)MROWS * INNER * 2);
    float*          dtb     = (float*)alloc((size_t)MROWS * INNER * 4);
    float*          Bm      = (float*)alloc((size_t)MROWS * STATE * 4);
    float*          Cm      = (float*)alloc((size_t)MROWS * STATE * 4);
    float*          yb      = (float*)alloc((size_t)MROWS * INNER * 4);
    unsigned short* z_bf    = (unsigned short*)alloc((size_t)MROWS * INNER * 2);

    auto cvt = [&](const float* s, unsigned short* d, int n) {
        cvt_bf16<<<(n + 255) / 256, 256, 0, stream>>>(s, d, n);
    };
    cvt(x,       x_bf,    MROWS * DIMX);
    cvt(in_proj, wip_bf,  NPROJ * DIMX);
    cvt(dt_w,    wdt_bf,  INNER * INNER);
    cvt(out_w,   wout_bf, DIMX * INNER);
    cvt(B_w,     wB_bf,   STATE * INNER);
    cvt(C_w,     wC_bf,   STATE * INNER);

    // 1) in_proj: xr[8192,4096] = x[8192,1024] @ in_proj_w[4096,1024]^T
    gemm_bf16<<<dim3(NPROJ / BN, MROWS / BM), 256, 0, stream>>>(
        x_bf, wip_bf, xr, nullptr, NPROJ, DIMX, 0);

    // 2) depthwise conv + SiLU -> x_in (f32 + bf16)
    {
        long long n = (long long)MROWS * INNER;
        conv_silu<<<(unsigned)((n + 255) / 256), 256, 0, stream>>>(
            xr, conv_w, conv_b, xin, xin_bf);
    }

    // 3) dt = softplus(x_in @ dt_w^T + dt_b)
    gemm_bf16<<<dim3(INNER / BN, MROWS / BM), 256, 0, stream>>>(
        xin_bf, wdt_bf, dtb, dt_b, INNER, INNER, 1);

    // 4) Bm, Cm = x_in @ {B_w,C_w}^T  (N=16 skinny WMMA)
    gemm_bc<<<dim3(MROWS / 128, 2), 256, 0, stream>>>(
        xin_bf, wB_bf, wC_bf, Bm, Cm, INNER);

    // 5) selective scan -> y
    scan_kernel<<<dim3(INNER / 256, BB), 256, 0, stream>>>(
        dtb, xin, Bm, Cm, A_log, yb);

    // 6) z = y * silu(res)  (bf16)
    {
        long long n = (long long)MROWS * INNER;
        zprep<<<(unsigned)((n + 255) / 256), 256, 0, stream>>>(yb, xr, z_bf);
    }

    // 7) out = z @ out_w^T
    gemm_bf16<<<dim3(DIMX / BN, MROWS / BM), 256, 0, stream>>>(
        z_bf, wout_bf, out, nullptr, DIMX, INNER, 0);
}
